// GCN_gosemsim_21328807592414
// MI455X (gfx1250) — compile-verified, matrix operand order checked by reading, not run on hardware
//
#include <hip/hip_runtime.h>
#include <hip/hip_bf16.h>

typedef _Float16 half_t;
typedef __attribute__((ext_vector_type(16))) _Float16 v16h;
typedef __attribute__((ext_vector_type(8)))  _Float16 v8h;
typedef __attribute__((ext_vector_type(8)))  float    v8f;

#define NGRAPH 128   // B*G
#define NNODE  512
#define NEDGE  8192
#define NFEAT  128
#define NHID   128
#define BN_EPS 1e-5f

// ---------------------------------------------------------------------------
// Stage 1: per-graph degree computation -> rsqrt norms
// ---------------------------------------------------------------------------
__global__ __launch_bounds__(512) void k_degrees(const int* __restrict__ edges,
                                                 float* __restrict__ rs_out,
                                                 float* __restrict__ rs_in) {
    __shared__ float degO[NNODE];
    __shared__ float degI[NNODE];
    const int g = blockIdx.x;
    const int t = threadIdx.x;
    const int* e = edges + (size_t)g * 2 * NEDGE;
    for (int n = t; n < NNODE; n += 512) { degO[n] = 0.f; degI[n] = 0.f; }
    __syncthreads();
    for (int i = t; i < NEDGE; i += 512) {
        atomicAdd(&degO[e[i]], 1.0f);          // out-degree over src
        atomicAdd(&degI[e[NEDGE + i]], 1.0f);  // in-degree over dst
    }
    __syncthreads();
    for (int n = t; n < NNODE; n += 512) {
        rs_out[g * NNODE + n] = rsqrtf(fmaxf(degO[n], 1.0f));
        rs_in [g * NNODE + n] = rsqrtf(fmaxf(degI[n], 1.0f));
    }
}

// ---------------------------------------------------------------------------
// Stage 2/4: edge scatter  agg[dst,:] += x[src,:] * rs_out[src]
// One edge per wave at a time; 128 features spread over 32 lanes x 4 chunks.
// ---------------------------------------------------------------------------
__global__ __launch_bounds__(512) void k_scatter(const float*  __restrict__ xf32,
                                                 const half_t* __restrict__ xf16,
                                                 const int*    __restrict__ edges,
                                                 const float*  __restrict__ rs_out,
                                                 float*        __restrict__ agg,
                                                 int use_half) {
    const int g    = blockIdx.x;
    const int wave = threadIdx.x >> 5;
    const int lane = threadIdx.x & 31;
    const int* e = edges + (size_t)g * 2 * NEDGE;
    const float* rso = rs_out + g * NNODE;
    const size_t gbase = (size_t)g * NNODE * NFEAT;
    for (int i = wave; i < NEDGE; i += 16) {
        const int src = e[i];
        const int dst = e[NEDGE + i];
        const float s = rso[src];
        const size_t sb = gbase + (size_t)src * NFEAT;
        const size_t db = gbase + (size_t)dst * NFEAT;
#pragma unroll
        for (int c = 0; c < 4; ++c) {
            const int f = lane + 32 * c;
            const float v = use_half ? (float)xf16[sb + f] : xf32[sb + f];
            atomicAdd(&agg[db + f], v * s);
        }
    }
}

// ---------------------------------------------------------------------------
// Stage 3/5: h = relu((agg * rs_in) @ W + b) via v_wmma_f32_16x16x32_f16.
// One M-tile per wave, 2 workgroups per graph -> the kernel body is fully
// straight-line (ko and ni completely unrolled): no loops survive, so LICM
// cannot hoist the 32 B fragments and spill them to scratch. The pressure-
// aware scheduler keeps B as just-in-time ds_load_b128 pairs.
// C[8] (64 VGPRs) + A (8) + in-flight B fits the (512,1) register budget.
// A layout (16-bit 16x32, ISA 7.12.2): lanes 0-15 row M=l16 carry K = kb+0..7
// and kb+16..23; lanes 16-31 carry K = kb+8..15 and kb+24..31.
// ---------------------------------------------------------------------------
__global__ __launch_bounds__(512, 1) void k_gemm(const float* __restrict__ agg,
                                                 const float* __restrict__ rs_in,
                                                 const float* __restrict__ W,  // [K][N] f32
                                                 const float* __restrict__ bvec,
                                                 half_t*      __restrict__ h_out,
                                                 float*       __restrict__ emb_out,
                                                 int layer) {
    __shared__ __align__(16) half_t wt[NHID * NHID];   // W^T, f16: wt[n*NHID + k]
    __shared__ float  bs[NHID];
    __shared__ float  embacc[NHID];
    const int g    = blockIdx.x >> 1;        // 2 blocks per graph
    const int half = blockIdx.x & 1;
    const int t    = threadIdx.x;

    for (int i = t; i < NHID * NHID; i += 512) {
        const int k = i >> 7, n = i & 127;
        wt[n * NHID + k] = (half_t)W[i];
    }
    if (t < NHID) { bs[t] = bvec[t]; embacc[t] = 0.f; }
    __syncthreads();

    const int wave = t >> 5, lane = t & 31;
    const int l16 = lane & 15, hi = lane >> 4;
    const size_t gb = (size_t)g * NNODE;

    const int mi = half * 16 + wave;          // 32 M-tiles over 2 blocks
    const int m  = mi * 16 + l16;
    const float* arow = agg + (gb + m) * NFEAT;
    const float  rsin = rs_in[gb + m];

    v8f c[8] = {};                            // 8 N-tiles resident in VGPRs

#pragma unroll
    for (int ko = 0; ko < 4; ++ko) {
        const int kb = ko * 32;
        const int k0 = kb + hi * 8;
        const int k1 = kb + 16 + hi * 8;
        const float4 f0 = *(const float4*)(arow + k0);
        const float4 f1 = *(const float4*)(arow + k0 + 4);
        const float4 f2 = *(const float4*)(arow + k1);
        const float4 f3 = *(const float4*)(arow + k1 + 4);
        v16h a;
        a[0]  = (half_t)(f0.x * rsin); a[1]  = (half_t)(f0.y * rsin);
        a[2]  = (half_t)(f0.z * rsin); a[3]  = (half_t)(f0.w * rsin);
        a[4]  = (half_t)(f1.x * rsin); a[5]  = (half_t)(f1.y * rsin);
        a[6]  = (half_t)(f1.z * rsin); a[7]  = (half_t)(f1.w * rsin);
        a[8]  = (half_t)(f2.x * rsin); a[9]  = (half_t)(f2.y * rsin);
        a[10] = (half_t)(f2.z * rsin); a[11] = (half_t)(f2.w * rsin);
        a[12] = (half_t)(f3.x * rsin); a[13] = (half_t)(f3.y * rsin);
        a[14] = (half_t)(f3.z * rsin); a[15] = (half_t)(f3.w * rsin);

#pragma unroll
        for (int ni = 0; ni < 8; ++ni) {
            const half_t* brow = wt + (ni * 16 + l16) * NHID;
            const v8h blo = *(const v8h*)(brow + k0);   // ds_load_b128
            const v8h bhi = *(const v8h*)(brow + k1);   // ds_load_b128
            const v16h b = __builtin_shufflevector(blo, bhi,
                               0, 1, 2, 3, 4, 5, 6, 7,
                               8, 9, 10, 11, 12, 13, 14, 15);
            c[ni] = __builtin_amdgcn_wmma_f32_16x16x32_f16(
                        false, a, false, b, (short)0, c[ni], false, false);
        }
    }

#pragma unroll
    for (int ni = 0; ni < 8; ++ni) {
        const int n = ni * 16 + l16;
        if (layer == 0) {
#pragma unroll
            for (int r = 0; r < 8; ++r) {
                const int ml = r + 8 * hi;                 // C layout: VGPR r
                const float v = fmaxf(c[ni][r] + bs[n], 0.f);
                h_out[(gb + (size_t)mi * 16 + ml) * NFEAT + n] = (half_t)v;
            }
        } else {
            float s = 0.f;
#pragma unroll
            for (int r = 0; r < 8; ++r) s += fmaxf(c[ni][r] + bs[n], 0.f);
            atomicAdd(&embacc[n], s);                      // mean-nodes fused
        }
    }

    if (layer == 1) {
        __syncthreads();
        // each of the 2 blocks contributes its partial mean into emb (zeroed)
        if (t < NHID) atomicAdd(&emb_out[g * NHID + t], embacc[t] * (1.0f / NNODE));
    }
}

// ---------------------------------------------------------------------------
// Stage 6: z = relu(BN(emb @ Wt + bt))   (2 MFLOP total — scalar is fine)
// ---------------------------------------------------------------------------
__global__ __launch_bounds__(128) void k_head(const float* __restrict__ emb,
                                              const float* __restrict__ Wt,
                                              const float* __restrict__ bt,
                                              const float* __restrict__ gamma,
                                              const float* __restrict__ beta,
                                              const float* __restrict__ rmean,
                                              const float* __restrict__ rvar,
                                              float* __restrict__ out) {
    __shared__ float er[NHID];
    const int g = blockIdx.x, n = threadIdx.x;
    er[n] = emb[g * NHID + n];
    __syncthreads();
    float z = bt[n];
    for (int k = 0; k < NHID; ++k) z += er[k] * Wt[k * NHID + n];
    z = gamma[n] * (z - rmean[n]) * rsqrtf(rvar[n] + BN_EPS) + beta[n];
    out[g * NHID + n] = fmaxf(z, 0.f);
}

// ---------------------------------------------------------------------------
extern "C" void kernel_launch(void* const* d_in, const int* in_sizes, int n_in,
                              void* d_out, int out_size, void* d_ws, size_t ws_size,
                              hipStream_t stream) {
    (void)in_sizes; (void)n_in; (void)out_size; (void)ws_size;
    const float* feats = (const float*)d_in[0];
    const int*   edges = (const int*)d_in[1];
    const float* W0    = (const float*)d_in[2];
    const float* b0    = (const float*)d_in[3];
    const float* W1    = (const float*)d_in[4];
    const float* b1    = (const float*)d_in[5];
    const float* Wt    = (const float*)d_in[6];
    const float* bt    = (const float*)d_in[7];
    const float* gamma = (const float*)d_in[8];
    const float* beta  = (const float*)d_in[9];
    const float* rmean = (const float*)d_in[10];
    const float* rvar  = (const float*)d_in[11];
    float* out = (float*)d_out;

    // workspace carve-up (all offsets 256B-aligned by construction)
    char* ws = (char*)d_ws;
    float*  rs_out = (float*)ws;  ws += (size_t)NGRAPH * NNODE * sizeof(float);
    float*  rs_in  = (float*)ws;  ws += (size_t)NGRAPH * NNODE * sizeof(float);
    float*  agg    = (float*)ws;  ws += (size_t)NGRAPH * NNODE * NFEAT * sizeof(float);
    half_t* h1     = (half_t*)ws; ws += (size_t)NGRAPH * NNODE * NFEAT * sizeof(half_t);
    float*  emb    = (float*)ws;  ws += (size_t)NGRAPH * NHID * sizeof(float);
    const size_t agg_bytes = (size_t)NGRAPH * NNODE * NFEAT * sizeof(float);
    const size_t emb_bytes = (size_t)NGRAPH * NHID * sizeof(float);

    hipMemsetAsync(agg, 0, agg_bytes, stream);
    k_degrees<<<NGRAPH, 512, 0, stream>>>(edges, rs_out, rs_in);

    // layer 0
    k_scatter<<<NGRAPH, 512, 0, stream>>>(feats, nullptr, edges, rs_out, agg, 0);
    k_gemm   <<<NGRAPH * 2, 512, 0, stream>>>(agg, rs_in, W0, b0, h1, emb, 0);

    // layer 1 (+ fused mean-nodes readout into zeroed emb)
    hipMemsetAsync(agg, 0, agg_bytes, stream);
    hipMemsetAsync(emb, 0, emb_bytes, stream);
    k_scatter<<<NGRAPH, 512, 0, stream>>>(nullptr, h1, edges, rs_out, agg, 1);
    k_gemm   <<<NGRAPH * 2, 512, 0, stream>>>(agg, rs_in, W1, b1, h1, emb, 1);

    // head: linear + BN(eval) + relu
    k_head<<<NGRAPH, 128, 0, stream>>>(emb, Wt, bt, gamma, beta, rmean, rvar, out);
}
